// CodecTransformerBlock_3393024164128
// MI455X (gfx1250) — compile-verified, matrix operand order checked by reading, not run on hardware
//
#include <hip/hip_runtime.h>
#include <hip/hip_bf16.h>

typedef __bf16 bf16;
typedef __attribute__((ext_vector_type(16))) __bf16 v16bf;
typedef __attribute__((ext_vector_type(8)))  __bf16 v8bf;
typedef __attribute__((ext_vector_type(8)))  float  v8f;

#define B_    2
#define T_    2048
#define DIM_  1024
#define HID_  4096
#define H_    16
#define KVH_  4
#define HD_   64
#define WIN_  512
#define MROWS_ 4096  /* B*T */

// ---------------------------------------------------------------------------
// Async global->LDS copy (CDNA5 GLOBAL_LOAD_ASYNC_TO_LDS_B128, ASYNCcnt).
// Builtin signature (from hipcc diagnostic): param0 = v4i addrspace(1)*,
// param1 = v4i addrspace(3)*, then imm offset, imm cpol.
// Falls back to a synchronous reg-staged copy when the builtin is absent.
// ---------------------------------------------------------------------------
#if defined(__has_builtin)
#if __has_builtin(__builtin_amdgcn_global_load_async_to_lds_b128)
#define HAVE_ASYNC_COPY 1
#endif
#endif

typedef int v4i __attribute__((vector_size(16)));
typedef __attribute__((address_space(1))) v4i gv4i;
typedef __attribute__((address_space(3))) v4i lv4i;

__device__ inline void async_cp16(const bf16* g, bf16* l) {
#ifdef HAVE_ASYNC_COPY
  __builtin_amdgcn_global_load_async_to_lds_b128((gv4i*)g, (lv4i*)l, 0, 0);
#else
  *(v8bf*)l = *(const v8bf*)g;
#endif
}

__device__ inline void async_wait0() {
#ifdef HAVE_ASYNC_COPY
#if __has_builtin(__builtin_amdgcn_s_wait_asynccnt)
  __builtin_amdgcn_s_wait_asynccnt(0);
#else
  asm volatile("s_wait_asynccnt 0x0" ::: "memory");
#endif
#endif
}

// ---------------------------------------------------------------------------
// WMMA helpers (CDNA5 16x16x32 bf16, fp32 accumulate)
// A-fragment (16x32 MxK): lane<16 -> M=lane, K elems {0..7,16..23};
// lane>=16 -> M=lane-16, K elems {8..15,24..31}. B mirrored with N on lanes.
// ---------------------------------------------------------------------------
__device__ inline v16bf frag_ld(const bf16* rowp, int half) {
  v8bf lo = *(const v8bf*)(rowp + half * 8);
  v8bf hi = *(const v8bf*)(rowp + 16 + half * 8);
  v16bf r;
#pragma unroll
  for (int i = 0; i < 8; ++i) { r[i] = lo[i]; r[i + 8] = hi[i]; }
  return r;
}

__device__ inline v8f wmma_bf16(v16bf a, v16bf b, v8f c) {
  return __builtin_amdgcn_wmma_f32_16x16x32_bf16(false, a, false, b, (short)0,
                                                 c, false, false);
}

// ---------------------------------------------------------------------------
// fp32 -> bf16 conversion
// ---------------------------------------------------------------------------
__global__ __launch_bounds__(256) void k_cvt(const float* __restrict__ in,
                                             bf16* __restrict__ out, size_t n) {
  size_t i = (size_t)blockIdx.x * blockDim.x + threadIdx.x;
  size_t stride = (size_t)gridDim.x * blockDim.x;
  for (; i < n; i += stride) out[i] = (bf16)in[i];
}

// ---------------------------------------------------------------------------
// Row RMSNorm (row length = cols), fp32 in, bf16 out. One block per row.
// ---------------------------------------------------------------------------
__global__ __launch_bounds__(256) void k_rmsnorm(const float* __restrict__ x,
                                                 const float* __restrict__ w,
                                                 bf16* __restrict__ out,
                                                 int cols, float eps) {
  __shared__ float red[8];
  const int tid = threadIdx.x;
  const int lane = tid & 31, wave = tid >> 5;
  const float* xr = x + (size_t)blockIdx.x * cols;
  float ss = 0.f;
  for (int c = tid; c < cols; c += 256) { float v = xr[c]; ss += v * v; }
#pragma unroll
  for (int off = 1; off < 32; off <<= 1) ss += __shfl_xor(ss, off, 32);
  if (lane == 0) red[wave] = ss;
  __syncthreads();
  float tot = 0.f;
#pragma unroll
  for (int i = 0; i < 8; ++i) tot += red[i];
  const float sc = rsqrtf(tot / (float)cols + eps);
  bf16* orow = out + (size_t)blockIdx.x * cols;
  for (int c = tid; c < cols; c += 256) orow[c] = (bf16)(xr[c] * sc * w[c]);
}

// ---------------------------------------------------------------------------
// Per-head q/k RMSNorm over HD=64, fp32 in, bf16 out. 8 rows per 256-blk.
// ---------------------------------------------------------------------------
__global__ __launch_bounds__(256) void k_qknorm(const float* __restrict__ x,
                                                const float* __restrict__ w,
                                                bf16* __restrict__ out,
                                                int rows, float eps) {
  const int lane = threadIdx.x & 31, wave = threadIdx.x >> 5;
  const int row = blockIdx.x * 8 + wave;
  if (row >= rows) return;
  const float* xr = x + (size_t)row * HD_;
  const float v0 = xr[lane], v1 = xr[lane + 32];
  float ss = v0 * v0 + v1 * v1;
#pragma unroll
  for (int off = 1; off < 32; off <<= 1) ss += __shfl_xor(ss, off, 32);
  const float sc = rsqrtf(ss / (float)HD_ + eps);
  bf16* orow = out + (size_t)row * HD_;
  orow[lane] = (bf16)(v0 * sc * w[lane]);
  orow[lane + 32] = (bf16)(v1 * sc * w[lane + 32]);
}

// ---------------------------------------------------------------------------
// SiLU gate: g = silu(u1) * u3, bf16 in/out
// ---------------------------------------------------------------------------
__global__ __launch_bounds__(256) void k_silu_mul(const bf16* __restrict__ u1,
                                                  const bf16* __restrict__ u3,
                                                  bf16* __restrict__ g,
                                                  size_t n) {
  size_t i = (size_t)blockIdx.x * blockDim.x + threadIdx.x;
  size_t stride = (size_t)gridDim.x * blockDim.x;
  for (; i < n; i += stride) {
    float a = (float)u1[i];
    float c = (float)u3[i];
    float s = a / (1.f + __expf(-a));
    g[i] = (bf16)(s * c);
  }
}

// ---------------------------------------------------------------------------
// Tiled WMMA GEMM: C(MxN) = A(MxK,bf16 row-major) * W(NxK,bf16 row-major)^T
// Block 128x256, 8 waves (2x4), each wave a 64x64 tile (4x4 16x16 tiles),
// K step 32, double-buffered LDS filled by async global->LDS copies.
// epi: 0 = fp32 store, 1 = bf16 store, 2 = Cf = resid + acc*scale[n].
// Requires M % 128 == 0, N % 256 == 0, K % 32 == 0 (true for all launches).
// ---------------------------------------------------------------------------
__global__ __launch_bounds__(256) void k_gemm(
    const bf16* __restrict__ A, const bf16* __restrict__ W, int M, int N,
    int K, float* __restrict__ Cf, bf16* __restrict__ Cb,
    const float* __restrict__ resid, const float* __restrict__ scale,
    int epi) {
  (void)M;
  __shared__ bf16 As[2][128][40];  // 40*2B = 80B rows -> 16B aligned frags
  __shared__ bf16 Bs[2][256][40];
  const int tid = threadIdx.x;
  const int lane = tid & 31, wave = tid >> 5;
  const int half = lane >> 4, lm = lane & 15;
  const int wm = wave & 1, wn = wave >> 1;
  const long mbase = (long)blockIdx.y * 128;
  const long nbase = (long)blockIdx.x * 256;

  v8f acc[4][4];
#pragma unroll
  for (int mt = 0; mt < 4; ++mt) {
#pragma unroll
    for (int nt = 0; nt < 4; ++nt) {
#pragma unroll
      for (int i = 0; i < 8; ++i) acc[mt][nt][i] = 0.f;
    }
  }

  const int nk = K >> 5;
  const int lr = tid >> 2, lc = (tid & 3) * 8;  // 64 rows x 4 colgroups

  auto copy_tile = [&](int buf, int kt) {
    const long kb = (long)kt * 32;
    const bf16* Ag = A + (mbase + lr) * K + kb + lc;
    async_cp16(Ag, &As[buf][lr][lc]);
    async_cp16(Ag + (long)64 * K, &As[buf][lr + 64][lc]);
    const bf16* Bg = W + (nbase + lr) * K + kb + lc;
#pragma unroll
    for (int i = 0; i < 4; ++i)
      async_cp16(Bg + (long)i * 64 * K, &Bs[buf][lr + i * 64][lc]);
  };

  copy_tile(0, 0);
  async_wait0();
  __syncthreads();

  for (int kt = 0; kt < nk; ++kt) {
    const int cur = kt & 1;
    if (kt + 1 < nk) copy_tile(cur ^ 1, kt + 1);  // overlaps with WMMAs below
    if (kt + 2 < nk) {  // global_prefetch_b8 hint two tiles ahead
      __builtin_prefetch(A + (mbase + lr) * K + (long)(kt + 2) * 32 + lc, 0, 1);
      __builtin_prefetch(W + (nbase + lr) * K + (long)(kt + 2) * 32 + lc, 0, 1);
    }
    v16bf af[4], bw[4];
#pragma unroll
    for (int mt = 0; mt < 4; ++mt)
      af[mt] = frag_ld(&As[cur][wm * 64 + mt * 16 + lm][0], half);
#pragma unroll
    for (int nt = 0; nt < 4; ++nt)
      bw[nt] = frag_ld(&Bs[cur][wn * 64 + nt * 16 + lm][0], half);
#pragma unroll
    for (int mt = 0; mt < 4; ++mt) {
#pragma unroll
      for (int nt = 0; nt < 4; ++nt)
        acc[mt][nt] = wmma_bf16(af[mt], bw[nt], acc[mt][nt]);
    }
    if (kt + 1 < nk) async_wait0();
    __syncthreads();
  }

  // C/D layout: lane<16 -> col N=lm, rows half*8+r (ISA C-tile table)
#pragma unroll
  for (int mt = 0; mt < 4; ++mt) {
#pragma unroll
    for (int nt = 0; nt < 4; ++nt) {
      const long n = nbase + wn * 64 + nt * 16 + lm;
#pragma unroll
      for (int r = 0; r < 8; ++r) {
        const long m = mbase + wm * 64 + mt * 16 + half * 8 + r;
        const size_t off = (size_t)m * N + n;
        const float vv = acc[mt][nt][r];
        if (epi == 0)
          Cf[off] = vv;
        else if (epi == 1)
          Cb[off] = (bf16)vv;
        else
          Cf[off] = resid[off] + vv * scale[n];
      }
    }
  }
}

// ---------------------------------------------------------------------------
// Sliding-window causal GQA flash attention. One wave per (b, h, 16-query
// tile). Keys processed in chunks of 32 staged in LDS via async copies.
// q: (B,T,H,HD) bf16 (post qk-norm); k,v: (B,T,KVH,HD) bf16; y: (B,T,H*HD).
// ---------------------------------------------------------------------------
__global__ __launch_bounds__(32) void k_attn(const bf16* __restrict__ q,
                                             const bf16* __restrict__ k,
                                             const bf16* __restrict__ v,
                                             bf16* __restrict__ y) {
  __shared__ bf16 Ks[32][72];  // key-major, 144B rows (16B aligned)
  __shared__ bf16 Vs[32][72];
  __shared__ bf16 Ps[16][40];  // P tile staging for A-fragment re-layout
  const int lane = threadIdx.x;
  const int half = lane >> 4, lm = lane & 15;
  const int qt = blockIdx.x, h = blockIdx.y, b = blockIdx.z;
  const int qbase = qt * 16;
  const int kvh = h / (H_ / KVH_);
  const float NEGB = -1e30f;
  const float SCL = 0.125f;  // 1/sqrt(HD)

  v16bf aq[2];
  {
    const bf16* qrow = q + (size_t)(b * T_ + qbase + lm) * (H_ * HD_) + h * HD_;
    aq[0] = frag_ld(qrow, half);
    aq[1] = frag_ld(qrow + 32, half);
  }

  v8f o[4];
  float mrow[8], lrow[8];
#pragma unroll
  for (int d = 0; d < 4; ++d) {
#pragma unroll
    for (int r = 0; r < 8; ++r) o[d][r] = 0.f;
  }
#pragma unroll
  for (int r = 0; r < 8; ++r) { mrow[r] = NEGB; lrow[r] = 0.f; }

  int lo = qbase - (WIN_ - 1);
  if (lo < 0) lo = 0;
  const int kc_lo = lo & ~31;
  const int kend = qbase + 15;

  for (int kc = kc_lo; kc <= kend; kc += 32) {
    {  // each lane stages one K row and one V row (64 bf16 each) into LDS
      const bf16* krow =
          k + (size_t)(b * T_ + kc + lane) * (KVH_ * HD_) + kvh * HD_;
      const bf16* vrow =
          v + (size_t)(b * T_ + kc + lane) * (KVH_ * HD_) + kvh * HD_;
#pragma unroll
      for (int j = 0; j < 8; ++j) {
        async_cp16(krow + j * 8, &Ks[lane][j * 8]);
        async_cp16(vrow + j * 8, &Vs[lane][j * 8]);
      }
    }
    async_wait0();
    __syncthreads();

    // S = Q * K^T : 2 n-tiles (16 keys each) x 2 k-steps (hd halves)
    v8f s[2];
#pragma unroll
    for (int nt = 0; nt < 2; ++nt) {
#pragma unroll
      for (int i = 0; i < 8; ++i) s[nt][i] = 0.f;
#pragma unroll
      for (int ks = 0; ks < 2; ++ks) {
        v16bf bk = frag_ld(&Ks[nt * 16 + lm][ks * 32], half);
        s[nt] = wmma_bf16(aq[ks], bk, s[nt]);
      }
    }

    // mask + online softmax (row stats replicated across each 16-lane half)
    float alpha8[8];
#pragma unroll
    for (int r = 0; r < 8; ++r) {
      const int qi = qbase + half * 8 + r;
      const int kj0 = kc + lm, kj1 = kc + 16 + lm;
      const bool ok0 = (kj0 <= qi) && (qi - kj0 < WIN_);
      const bool ok1 = (kj1 <= qi) && (qi - kj1 < WIN_);
      float s0 = ok0 ? s[0][r] * SCL : NEGB;
      float s1 = ok1 ? s[1][r] * SCL : NEGB;
      float rm = fmaxf(s0, s1);
#pragma unroll
      for (int off = 1; off < 16; off <<= 1)
        rm = fmaxf(rm, __shfl_xor(rm, off, 32));
      const float mnew = fmaxf(mrow[r], rm);
      const float al = __expf(mrow[r] - mnew);
      const float p0 = __expf(s0 - mnew);
      const float p1 = __expf(s1 - mnew);
      float ps = p0 + p1;
#pragma unroll
      for (int off = 1; off < 16; off <<= 1) ps += __shfl_xor(ps, off, 32);
      lrow[r] = lrow[r] * al + ps;
      mrow[r] = mnew;
      alpha8[r] = al;
      Ps[half * 8 + r][lm] = (bf16)p0;
      Ps[half * 8 + r][16 + lm] = (bf16)p1;
    }
    __syncthreads();

#pragma unroll
    for (int d = 0; d < 4; ++d) {
#pragma unroll
      for (int r = 0; r < 8; ++r) o[d][r] *= alpha8[r];
    }

    // PV: A = P (16x32), B = V chunk (32 keys x 64 hd), 4 hd n-tiles
    v16bf pa = frag_ld(&Ps[lm][0], half);
#pragma unroll
    for (int d = 0; d < 4; ++d) {
      v16bf bv;
#pragma unroll
      for (int i = 0; i < 8; ++i) {
        bv[i] = Vs[half * 8 + i][d * 16 + lm];
        bv[i + 8] = Vs[16 + half * 8 + i][d * 16 + lm];
      }
      o[d] = wmma_bf16(pa, bv, o[d]);
    }
    __syncthreads();
  }

#pragma unroll
  for (int d = 0; d < 4; ++d) {
#pragma unroll
    for (int r = 0; r < 8; ++r) {
      const int qi = qbase + half * 8 + r;
      const float val = o[d][r] / lrow[r];
      y[(size_t)(b * T_ + qi) * (H_ * HD_) + h * HD_ + d * 16 + lm] = (bf16)val;
    }
  }
}

// ---------------------------------------------------------------------------
// Host orchestration
// ---------------------------------------------------------------------------
extern "C" void kernel_launch(void* const* d_in, const int* in_sizes, int n_in,
                              void* d_out, int out_size, void* d_ws,
                              size_t ws_size, hipStream_t stream) {
  (void)in_sizes; (void)n_in; (void)out_size; (void)ws_size;
  const float* x           = (const float*)d_in[0];
  const float* wq          = (const float*)d_in[1];
  const float* wk          = (const float*)d_in[2];
  const float* wv          = (const float*)d_in[3];
  const float* wo          = (const float*)d_in[4];
  const float* w1          = (const float*)d_in[5];
  const float* w2          = (const float*)d_in[6];
  const float* w3          = (const float*)d_in[7];
  const float* q_norm_w    = (const float*)d_in[8];
  const float* k_norm_w    = (const float*)d_in[9];
  const float* attn_norm_w = (const float*)d_in[10];
  const float* ffn_norm_w  = (const float*)d_in[11];
  const float* attn_scale  = (const float*)d_in[12];
  const float* ffn_scale   = (const float*)d_in[13];

  char* base = (char*)d_ws;
  size_t off = 0;
  auto alloc = [&](size_t bytes) -> void* {
    void* p = base + off;
    off += (bytes + 255) & ~(size_t)255;
    return p;
  };

  const size_t M = MROWS_;
  bf16* wq_b = (bf16*)alloc((size_t)H_ * HD_ * DIM_ * 2);
  bf16* wk_b = (bf16*)alloc((size_t)KVH_ * HD_ * DIM_ * 2);
  bf16* wv_b = (bf16*)alloc((size_t)KVH_ * HD_ * DIM_ * 2);
  bf16* wo_b = (bf16*)alloc((size_t)DIM_ * H_ * HD_ * 2);
  bf16* w1_b = (bf16*)alloc((size_t)HID_ * DIM_ * 2);
  bf16* w2_b = (bf16*)alloc((size_t)DIM_ * HID_ * 2);
  bf16* w3_b = (bf16*)alloc((size_t)HID_ * DIM_ * 2);
  bf16* xn_b = (bf16*)alloc(M * DIM_ * 2);
  float* qf  = (float*)alloc(M * (size_t)(H_ * HD_) * 4);
  float* kf  = (float*)alloc(M * (size_t)(KVH_ * HD_) * 4);
  bf16* qb   = (bf16*)alloc(M * (size_t)(H_ * HD_) * 2);
  bf16* kb   = (bf16*)alloc(M * (size_t)(KVH_ * HD_) * 2);
  bf16* vb   = (bf16*)alloc(M * (size_t)(KVH_ * HD_) * 2);
  bf16* yb   = (bf16*)alloc(M * (size_t)(H_ * HD_) * 2);
  float* hf  = (float*)alloc(M * DIM_ * 4);
  bf16* hn_b = (bf16*)alloc(M * DIM_ * 2);
  bf16* u1_b = (bf16*)alloc(M * (size_t)HID_ * 2);
  bf16* u3_b = (bf16*)alloc(M * (size_t)HID_ * 2);
  bf16* g_b  = (bf16*)alloc(M * (size_t)HID_ * 2);

  auto cvt = [&](const float* src, bf16* dst, size_t n) {
    int blocks = (int)((n + 1023) / 1024);
    k_cvt<<<blocks, 256, 0, stream>>>(src, dst, n);
  };
  cvt(wq, wq_b, (size_t)H_ * HD_ * DIM_);
  cvt(wk, wk_b, (size_t)KVH_ * HD_ * DIM_);
  cvt(wv, wv_b, (size_t)KVH_ * HD_ * DIM_);
  cvt(wo, wo_b, (size_t)DIM_ * H_ * HD_);
  cvt(w1, w1_b, (size_t)HID_ * DIM_);
  cvt(w2, w2_b, (size_t)DIM_ * HID_);
  cvt(w3, w3_b, (size_t)HID_ * DIM_);

  // xn = rmsnorm(x, attn_norm_w)
  k_rmsnorm<<<(int)M, 256, 0, stream>>>(x, attn_norm_w, xn_b, DIM_, 1e-5f);

  // q,k (fp32 for qk-norm), v directly bf16
  k_gemm<<<dim3((H_ * HD_) / 256, (int)M / 128), 256, 0, stream>>>(
      xn_b, wq_b, (int)M, H_ * HD_, DIM_, qf, nullptr, nullptr, nullptr, 0);
  k_gemm<<<dim3((KVH_ * HD_) / 256, (int)M / 128), 256, 0, stream>>>(
      xn_b, wk_b, (int)M, KVH_ * HD_, DIM_, kf, nullptr, nullptr, nullptr, 0);
  k_gemm<<<dim3((KVH_ * HD_) / 256, (int)M / 128), 256, 0, stream>>>(
      xn_b, wv_b, (int)M, KVH_ * HD_, DIM_, nullptr, vb, nullptr, nullptr, 1);

  k_qknorm<<<(int)(M * H_ / 8), 256, 0, stream>>>(qf, q_norm_w, qb,
                                                  (int)(M * H_), 1e-6f);
  k_qknorm<<<(int)(M * KVH_ / 8), 256, 0, stream>>>(kf, k_norm_w, kb,
                                                    (int)(M * KVH_), 1e-6f);

  // attention -> yb
  k_attn<<<dim3(T_ / 16, H_, B_), 32, 0, stream>>>(qb, kb, vb, yb);

  // h = x + (y @ wo^T) * attn_scale
  k_gemm<<<dim3(DIM_ / 256, (int)M / 128), 256, 0, stream>>>(
      yb, wo_b, (int)M, DIM_, H_ * HD_, hf, nullptr, x, attn_scale, 2);

  // hn = rmsnorm(h, ffn_norm_w)
  k_rmsnorm<<<(int)M, 256, 0, stream>>>(hf, ffn_norm_w, hn_b, DIM_, 1e-5f);

  // u1 = hn @ w1^T, u3 = hn @ w3^T (bf16 stores)
  k_gemm<<<dim3(HID_ / 256, (int)M / 128), 256, 0, stream>>>(
      hn_b, w1_b, (int)M, HID_, DIM_, nullptr, u1_b, nullptr, nullptr, 1);
  k_gemm<<<dim3(HID_ / 256, (int)M / 128), 256, 0, stream>>>(
      hn_b, w3_b, (int)M, HID_, DIM_, nullptr, u3_b, nullptr, nullptr, 1);

  // g = silu(u1) * u3
  {
    size_t n = M * (size_t)HID_;
    int blocks = (int)((n + 1023) / 1024);
    k_silu_mul<<<blocks, 256, 0, stream>>>(u1_b, u3_b, g_b, n);
  }

  // out = h + (g @ w2^T) * ffn_scale
  k_gemm<<<dim3(DIM_ / 256, (int)M / 128), 256, 0, stream>>>(
      g_b, w2_b, (int)M, DIM_, HID_, (float*)d_out, nullptr, hf, ffn_scale, 2);
}
